// PhiAttention_81484119540369
// MI455X (gfx1250) — matrix operand order repaired and checked by reading, not compiled, not run on hardware
//
#include <hip/hip_runtime.h>
#include <hip/hip_bf16.h>

// ---------------------------------------------------------------------------
// Phi attention block on gfx1250 (MI455X): f16 WMMA GEMMs + flash attention.
// Shapes: B=2, T=2048, D=2048, H=32, HD=64, ROT=32.
// GEMM B-operand staged via double-buffered GLOBAL_LOAD_ASYNC_TO_LDS; per-wave
// tile sized (16x64) to keep accumulators resident (no scratch spills).
// ---------------------------------------------------------------------------

typedef _Float16 f16;
typedef __attribute__((ext_vector_type(16))) _Float16 v16h;
typedef __attribute__((ext_vector_type(8)))  float    v8f;
typedef __attribute__((ext_vector_type(4)))  unsigned int u32x4;

union V16H { v16h v; u32x4 u[2]; f16 h[16]; };

#define WMMA_F16(Av, Bv, Cv) \
    __builtin_amdgcn_wmma_f32_16x16x32_f16(false, (Av), false, (Bv), (short)0, (Cv), false, false)

static constexpr int Bsz = 2;
static constexpr int T   = 2048;
static constexpr int D   = 2048;
static constexpr int H   = 32;
static constexpr int HD  = 64;

// LDS B-tile row stride in halves: 128 data + 8 pad -> 68 dwords/row, so the
// 32 K-lanes of a ds_load_b128 spread across all 64 banks (2 dwords/bank).
static constexpr int BSTRIDE = 136;
static constexpr int KSTEP   = 64;                 // k-depth per stage
static constexpr int BTILE   = KSTEP * BSTRIDE;    // halves per buffer

// -------------------------------- converts ---------------------------------

__global__ void cvt_f32_to_f16(const float* __restrict__ in, f16* __restrict__ out, int n) {
    int i = blockIdx.x * 256 + threadIdx.x;
    if (i < n) out[i] = (f16)in[i];
}

// Wt[k*N + n] = W[n*K + k]  (transpose so WMMA B-operand rows are contiguous)
__global__ void cvtT_f32_to_f16(const float* __restrict__ W, f16* __restrict__ Wt) {
    int i = blockIdx.x * 256 + threadIdx.x;       // i indexes Wt
    if (i < D * D) {
        int k = i / D;
        int n = i % D;
        Wt[i] = (f16)W[(size_t)n * D + k];
    }
}

// ------------------------------ WMMA GEMM ----------------------------------
// C[M=4096, N=2048] = Xh[M,K=2048] * Wt[K,N] + bias
// Block: 256 threads = 8 waves arranged 4(M) x 2(N); block tile 64x128.
// Per wave: 16x64 output (4 v8f accumulators). B panel (64x128 f16 per
// stage) staged in LDS by async copies, double buffered, shared by all
// waves. Epilogue mode: 0=Qh [b,h,t,d], 1=Kt [b,h,d,t], 2=Vh [b,h,t,d],
// 3=float out [m,n].
__global__ void __launch_bounds__(256)
gemm_wmma_kernel(const f16* __restrict__ X, const f16* __restrict__ Wt,
                 const float* __restrict__ bias,
                 f16* __restrict__ outH, float* __restrict__ outF, int mode) {
    __shared__ f16 bstage[2][BTILE];

    const int K = 2048, N = 2048;
    int tid = threadIdx.x;
    int w   = tid >> 5;
    int wm  = w & 3;            // wave M position (0..3)
    int wn  = w >> 2;           // wave N position (0..1)
    int l   = tid & 31;
    int lm  = l & 15;
    int hi  = l >> 4;
    int m0  = blockIdx.x * 64 + wm * 16;
    int n0  = blockIdx.y * 128;
    int nw  = wn * 64;          // wave's column offset inside the 128-wide tile

    v8f acc[4];
#pragma unroll
    for (int t = 0; t < 4; ++t)
        acc[t] = (v8f){0.f, 0.f, 0.f, 0.f, 0.f, 0.f, 0.f, 0.f};

    const f16* arow = X + (size_t)(m0 + lm) * K;

    // LDS byte offset of the staging buffer (low 32 bits of the generic
    // shared-memory address are the wave-relative LDS offset).
    unsigned bbase = (unsigned)(uintptr_t)(&bstage[0][0]);

    // Each thread async-copies four b128 chunks (8 halves each) of the
    // 64x128 tile. chunk c: row = c>>4, col = (c&15)*8.
    auto stage = [&](int buf, int kbase) {
#pragma unroll
        for (int j = 0; j < 4; ++j) {
            int c   = tid * 4 + j;
            int row = c >> 4;
            int col = (c & 15) * 8;
            const f16* g = Wt + (size_t)(kbase + row) * N + n0 + col;
            unsigned lds = bbase + (unsigned)((buf * BTILE + row * BSTRIDE + col) * 2);
            asm volatile("global_load_async_to_lds_b128 %0, %1, off"
                         :: "v"(lds), "v"(g) : "memory");
        }
    };

    stage(0, 0);

    const int NSTAGE = K / KSTEP;   // 32
    for (int kk = 0; kk < NSTAGE; ++kk) {
        int k = kk * KSTEP;
        if (kk + 1 < NSTAGE) {
            stage((kk + 1) & 1, k + KSTEP);
            // In-order async completion: <=4 outstanding means the current
            // stage has fully landed in LDS.
            asm volatile("s_wait_asynccnt 0x4" ::: "memory");
        } else {
            asm volatile("s_wait_asynccnt 0x0" ::: "memory");
        }
        __syncthreads();   // other threads' async writes visible

        const f16* bb = &bstage[kk & 1][0];
#pragma unroll
        for (int kb = 0; kb < 2; ++kb) {
            // A operand: lane m = lm, halves follow documented K permutation.
            V16H a;
            a.u[0] = *(const u32x4*)(arow + k + kb * 32 + hi * 8);
            a.u[1] = *(const u32x4*)(arow + k + kb * 32 + 16 + hi * 8);

            const f16* brow = bb + kb * 32 * BSTRIDE + nw + lm;   // B lane = N column, elements = K (table order)
#pragma unroll
            for (int t = 0; t < 4; ++t) {
                V16H b;
#pragma unroll
                for (int s = 0; s < 16; ++s) {
                    int kr = (s < 8) ? (hi * 8 + s) : (16 + hi * 8 + (s - 8));
                    b.h[s] = brow[kr * BSTRIDE + t * 16];
                }
                acc[t] = WMMA_F16(a.v, b.v, acc[t]);
            }
        }
        __builtin_prefetch(arow + k + 2 * KSTEP, 0, 0);
        __syncthreads();   // done reading this buffer before it is re-staged
    }

    // Epilogue: C layout -> component i is row (i + hi*8), col = lm (+ tile*16)
#pragma unroll
    for (int t = 0; t < 4; ++t) {
#pragma unroll
        for (int i = 0; i < 8; ++i) {
            int row = m0 + i + hi * 8;
            int n   = n0 + nw + t * 16 + lm;
            float v = acc[t][i] + bias[n];
            if (mode == 3) {
                outF[(size_t)row * N + n] = v;
            } else {
                int bb2 = row >> 11;         // row / T
                int tt  = row & 2047;        // row % T
                int hh  = n >> 6;            // n / HD
                int dd  = n & 63;            // n % HD
                size_t idx;
                if (mode == 1)  // Kt: [b,h,d,t]
                    idx = (((size_t)(bb2 * H + hh)) * HD + dd) * T + tt;
                else            // Qh / Vh: [b,h,t,d]
                    idx = (((size_t)(bb2 * H + hh)) * T + tt) * HD + dd;
                outH[idx] = (f16)v;
            }
        }
    }
}

// --------------------------------- RoPE ------------------------------------
// Q [b,h,t,d] and Kt [b,h,d,t] in place. ROT=32, pairs (d, d+16), d<16.
__global__ void rope_kernel(f16* __restrict__ Qh, f16* __restrict__ Kt) {
    int idx = blockIdx.x * 128 + threadIdx.x;   // total B*H*T*16
    int d  = idx & 15;
    int t  = (idx >> 4) & 2047;
    int bh = idx >> 15;
    if (bh >= Bsz * H) return;

    float freq = (float)t * __powf(10000.0f, -(float)d / 16.0f);
    float c = __cosf(freq), s = __sinf(freq);

    size_t qb = ((size_t)bh * T + t) * HD;
    float q0 = (float)Qh[qb + d], q1 = (float)Qh[qb + 16 + d];
    Qh[qb + d]      = (f16)(q0 * c - q1 * s);
    Qh[qb + 16 + d] = (f16)(q1 * c + q0 * s);

    size_t k0i = ((size_t)bh * HD + d) * T + t;
    size_t k1i = ((size_t)bh * HD + 16 + d) * T + t;
    float k0 = (float)Kt[k0i], k1 = (float)Kt[k1i];
    Kt[k0i] = (f16)(k0 * c - k1 * s);
    Kt[k1i] = (f16)(k1 * c + k0 * s);
}

// ----------------------------- Flash attention -----------------------------
// One wave per 16-row q tile; block = 8 waves = 128 q rows. Causal, online
// softmax. S and PV via v_wmma_f32_16x16x32_f16; P re-layout via LDS.
__global__ void __launch_bounds__(256)
flash_attn_kernel(const f16* __restrict__ Qh, const f16* __restrict__ Kt,
                  const f16* __restrict__ Vh, f16* __restrict__ AoH) {
    __shared__ f16 pbuf[8][16][32];   // per-wave P tile (16 rows x 32 keys)

    int w  = threadIdx.x >> 5;
    int l  = threadIdx.x & 31;
    int lm = l & 15;
    int hi = l >> 4;

    int bh    = blockIdx.x >> 4;        // 16 q-blocks of 128 per (b,h)
    int qblk  = blockIdx.x & 15;
    int qbase = qblk * 128 + w * 16;
    int b     = bh >> 5;
    int h     = bh & 31;

    // Q A-operands (d = 0..31 and 32..63), documented A K-permutation
    const f16* qp = Qh + ((size_t)bh * T + qbase + lm) * HD;
    V16H aQ[2];
#pragma unroll
    for (int kb = 0; kb < 2; ++kb) {
        aQ[kb].u[0] = *(const u32x4*)(qp + kb * 32 + hi * 8);
        aQ[kb].u[1] = *(const u32x4*)(qp + kb * 32 + 16 + hi * 8);
    }

    v8f o[4];
#pragma unroll
    for (int t4 = 0; t4 < 4; ++t4)
        o[t4] = (v8f){0.f, 0.f, 0.f, 0.f, 0.f, 0.f, 0.f, 0.f};
    float mr[8], ls[8];
#pragma unroll
    for (int i = 0; i < 8; ++i) { mr[i] = -3.0e38f; ls[i] = 0.f; }

    for (int jb = 0; jb <= qbase + 15; jb += 32) {
        // ---- S = Q * K^T for 16x32 key block (two 16-wide N tiles) ----
        v8f s[2];
#pragma unroll
        for (int t = 0; t < 2; ++t) {
            v8f c = (v8f){0.f, 0.f, 0.f, 0.f, 0.f, 0.f, 0.f, 0.f};
#pragma unroll
            for (int kb = 0; kb < 2; ++kb) {
                const f16* kp = Kt + ((size_t)bh * HD + kb * 32) * T + jb + t * 16 + lm;   // B lane = key, elements = d (table order)
                V16H bm;
#pragma unroll
                for (int s = 0; s < 16; ++s) {
                    int kr = (s < 8) ? (hi * 8 + s) : (16 + hi * 8 + (s - 8));
                    bm.h[s] = kp[(size_t)kr * T];
                }
                c = WMMA_F16(aQ[kb].v, bm.v, c);
            }
            s[t] = c;
        }

        // ---- scale, causal mask, online softmax (rows live in 16-lane groups)
        float alpha[8];
#pragma unroll
        for (int i = 0; i < 8; ++i) {
            int row = qbase + i + hi * 8;
            float s0 = s[0][i] * 0.125f;                 // 1/sqrt(64)
            float s1 = s[1][i] * 0.125f;
            if (jb + lm > row)      s0 = -3.0e38f;
            if (jb + 16 + lm > row) s1 = -3.0e38f;
            float mx = fmaxf(s0, s1);
            mx = fmaxf(mx, __shfl_xor(mx, 1, 32));
            mx = fmaxf(mx, __shfl_xor(mx, 2, 32));
            mx = fmaxf(mx, __shfl_xor(mx, 4, 32));
            mx = fmaxf(mx, __shfl_xor(mx, 8, 32));
            float mnew = fmaxf(mr[i], mx);
            alpha[i] = __expf(mr[i] - mnew);
            mr[i] = mnew;
            float p0 = __expf(s0 - mnew);
            float p1 = __expf(s1 - mnew);
            float rs = p0 + p1;
            rs += __shfl_xor(rs, 1, 32);
            rs += __shfl_xor(rs, 2, 32);
            rs += __shfl_xor(rs, 4, 32);
            rs += __shfl_xor(rs, 8, 32);
            ls[i] = ls[i] * alpha[i] + rs;
            pbuf[w][i + hi * 8][lm]      = (f16)p0;
            pbuf[w][i + hi * 8][16 + lm] = (f16)p1;
        }

        asm volatile("s_wait_dscnt 0x0" ::: "memory");   // wave-private LDS RAW

        // ---- P as A-operand (lane = row lm; key chunks per A layout) ----
        V16H aP;
        aP.u[0] = *(const u32x4*)(&pbuf[w][lm][hi * 8]);
        aP.u[1] = *(const u32x4*)(&pbuf[w][lm][16 + hi * 8]);

        // ---- O = O*alpha + P*V ----
#pragma unroll
        for (int t4 = 0; t4 < 4; ++t4) {
#pragma unroll
            for (int i = 0; i < 8; ++i) o[t4][i] *= alpha[i];
            const f16* vp = Vh + ((size_t)bh * T + jb) * HD + t4 * 16 + lm;
            V16H bv;
#pragma unroll
            for (int s = 0; s < 16; ++s) {
                int kr = (s < 8) ? (hi * 8 + s) : (16 + hi * 8 + (s - 8));
                bv.h[s] = vp[kr * HD];
            }
            o[t4] = WMMA_F16(aP.v, bv.v, o[t4]);
        }
    }

    // ---- normalize + store attention output in [M=b*T+t, D=h*HD+d] f16 ----
#pragma unroll
    for (int t4 = 0; t4 < 4; ++t4) {
#pragma unroll
        for (int i = 0; i < 8; ++i) {
            int row = qbase + i + hi * 8;
            int d   = t4 * 16 + lm;
            float v = o[t4][i] / ls[i];
            AoH[((size_t)b * T + row) * D + h * HD + d] = (f16)v;
        }
    }
}

// ------------------------------- launcher ----------------------------------

extern "C" void kernel_launch(void* const* d_in, const int* in_sizes, int n_in,
                              void* d_out, int out_size, void* d_ws, size_t ws_size,
                              hipStream_t stream) {
    const float* hs = (const float*)d_in[0];
    const float* Wq = (const float*)d_in[1];
    const float* bq = (const float*)d_in[2];
    const float* Wk = (const float*)d_in[3];
    const float* bk = (const float*)d_in[4];
    const float* Wv = (const float*)d_in[5];
    const float* bv = (const float*)d_in[6];
    const float* Wd = (const float*)d_in[7];
    const float* bd = (const float*)d_in[8];
    float* out = (float*)d_out;

    const size_t MX = (size_t)Bsz * T * D;   // 8388608
    const size_t WN = (size_t)D * D;         // 4194304

    f16* ws  = (f16*)d_ws;
    f16* Xh  = ws;
    f16* WtQ = Xh  + MX;
    f16* WtK = WtQ + WN;
    f16* WtV = WtK + WN;
    f16* WtD = WtV + WN;
    f16* Qh  = WtD + WN;
    f16* Kt  = Qh  + MX;
    f16* Vh  = Kt  + MX;
    f16* AoH = Vh  + MX;

    cvt_f32_to_f16<<<(int)(MX / 256), 256, 0, stream>>>(hs, Xh, (int)MX);
    cvtT_f32_to_f16<<<(int)(WN / 256), 256, 0, stream>>>(Wq, WtQ);
    cvtT_f32_to_f16<<<(int)(WN / 256), 256, 0, stream>>>(Wk, WtK);
    cvtT_f32_to_f16<<<(int)(WN / 256), 256, 0, stream>>>(Wv, WtV);
    cvtT_f32_to_f16<<<(int)(WN / 256), 256, 0, stream>>>(Wd, WtD);

    dim3 g(64, 16);   // M/64 x N/128
    gemm_wmma_kernel<<<g, 256, 0, stream>>>(Xh, WtQ, bq, Qh, nullptr, 0);
    gemm_wmma_kernel<<<g, 256, 0, stream>>>(Xh, WtK, bk, Kt, nullptr, 1);
    gemm_wmma_kernel<<<g, 256, 0, stream>>>(Xh, WtV, bv, Vh, nullptr, 2);

    rope_kernel<<<16384, 128, 0, stream>>>(Qh, Kt);

    flash_attn_kernel<<<Bsz * H * (T / 128), 256, 0, stream>>>(Qh, Kt, Vh, AoH);

    gemm_wmma_kernel<<<g, 256, 0, stream>>>(AoH, WtD, bd, nullptr, out, 3);
}
